// PositionAttentionModule_1382979469951
// MI455X (gfx1250) — matrix operand order repaired and checked by reading, not compiled, not run on hardware
//
#include <hip/hip_runtime.h>

typedef _Float16 v16h __attribute__((ext_vector_type(16)));
typedef _Float16 v8h  __attribute__((ext_vector_type(8)));
typedef float    v8f  __attribute__((ext_vector_type(8)));

#define BATCH 4
#define CIN   512
#define CQ    64
#define HW    4096
#define TILE  64

union Frag16 { v16h v; v8h h[2]; };

// A-matrix 16x32 f16 fragment: lane (m = lane%16, h = lane/16) holds
// K = ks + h*8 + [0..7]  and  K = ks + 16 + h*8 + [0..7]  of row m.
__device__ __forceinline__ v16h load_a_frag(const _Float16* row, int ks, int lh) {
  Frag16 u;
  u.h[0] = *(const v8h*)(row + ks + lh * 8);
  u.h[1] = *(const v8h*)(row + ks + 16 + lh * 8);
  return u.v;
}

// B-matrix 32x16 f16 fragment: lane (n = lane%16, h = lane/16) holds
// K = ks + h*8 + [0..7]  and  K = ks + 16 + h*8 + [0..7] of column n (column stored contiguously over K).
__device__ __forceinline__ v16h load_b_frag(const _Float16* col, int ks, int lh) {
  Frag16 u;
  u.h[0] = *(const v8h*)(col + ks + lh * 8);
  u.h[1] = *(const v8h*)(col + ks + 16 + lh * 8);
  return u.v;
}

// Async-copy one contiguous 8KB tile (64 rows x 128B) from global into a
// padded LDS tile [64][72] f16 (row pitch 144B) via the CDNA5 async-to-LDS
// path (ASYNCcnt-tracked, no VGPR staging). 256 threads x 2 x b128.
__device__ __forceinline__ void async_copy_tile64(const _Float16* gsrc,
                                                  _Float16 (*dst)[72], int tid) {
#pragma unroll
  for (int k = 0; k < 2; ++k) {
    const int e   = tid + k * 256;       // 16-byte chunk index (0..511)
    const int row = e >> 3;
    const int col = (e & 7) * 8;         // in halves
    const unsigned lds_off = (unsigned)(uintptr_t)(&dst[row][col]);
    const unsigned long long ga = (unsigned long long)(uintptr_t)(gsrc + e * 8);
    asm volatile("global_load_async_to_lds_b128 %0, %1, off"
                 :: "v"(lds_off), "v"(ga) : "memory");
  }
}

__device__ __forceinline__ void wait_async_all() {
  asm volatile("s_wait_asynccnt 0x0" ::: "memory");
}

// ---------------------------------------------------------------------------
// Kernel 1: fused 1x1-conv projections  y = W_cat @ x + b_cat  (f32 -> f16)
// Virtual M = 640 rows: [0,64)=Q, [64,128)=K, [128,640)=V.
// Q/K are stored TRANSPOSED: qT[b][pixel][c] so attention tiles are
// contiguous 8KB blocks (async-to-LDS friendly). V stays [b][c][pixel].
// ---------------------------------------------------------------------------
__global__ __launch_bounds__(256)
void qkv_proj_kernel(const float* __restrict__ x,
                     const float* __restrict__ Wq, const float* __restrict__ bq,
                     const float* __restrict__ Wk, const float* __restrict__ bk,
                     const float* __restrict__ Wv, const float* __restrict__ bv,
                     _Float16* __restrict__ qT, _Float16* __restrict__ kT,
                     _Float16* __restrict__ vh)
{
  __shared__ _Float16 LA[TILE][72];   // LA[o][c]
  __shared__ _Float16 LB[TILE][72];   // LB[n][c]  (x tile, transposed)

  const int n0   = blockIdx.x * TILE;
  const int mblk = blockIdx.y;
  const int b    = blockIdx.z;
  const int tid  = threadIdx.x;
  const int wave = tid >> 5;
  const int lane = tid & 31;
  const int lh   = lane >> 4;
  const int ll   = lane & 15;

  const float* Wsel; const float* bsel; int wrow0;
  if (mblk == 0)      { Wsel = Wq; bsel = bq; wrow0 = 0; }
  else if (mblk == 1) { Wsel = Wk; bsel = bk; wrow0 = 0; }
  else                { Wsel = Wv; bsel = bv; wrow0 = (mblk - 2) * TILE; }

  v8f acc[2];
#pragma unroll
  for (int it = 0; it < 2; ++it)
#pragma unroll
    for (int r = 0; r < 8; ++r) acc[it][r] = 0.0f;

  for (int kk = 0; kk < CIN; kk += TILE) {
    { // stage weights: LA[o][c] = W[wrow0+o][kk+c]
      const int o  = tid & 63;
      const int cb = (tid >> 6) * 16;
      const float* src = Wsel + (size_t)(wrow0 + o) * CIN + kk + cb;
#pragma unroll
      for (int ci = 0; ci < 16; ++ci) LA[o][cb + ci] = (_Float16)src[ci];
    }
    { // stage activations transposed: LB[n][c] = x[b][kk+c][n0+n]
      const int n  = tid & 63;
      const int cb = (tid >> 6) * 16;
#pragma unroll
      for (int ci = 0; ci < 16; ++ci) {
        const int c = cb + ci;
        LB[n][c] = (_Float16)x[((size_t)b * CIN + kk + c) * HW + n0 + n];
      }
    }
    __syncthreads();

#pragma unroll
    for (int it = 0; it < 2; ++it) {
      const int t  = wave * 2 + it;
      const int ti = t >> 2, tj = t & 3;
      const _Float16* arow = &LA[ti * 16 + ll][0];
      const _Float16* bcol = &LB[tj * 16 + ll][0];
#pragma unroll
      for (int ks = 0; ks < TILE; ks += 32) {
        v16h af = load_a_frag(arow, ks, lh);
        v16h bf = load_b_frag(bcol, ks, lh);
        acc[it] = __builtin_amdgcn_wmma_f32_16x16x32_f16(false, af, false, bf,
                                                         (short)0, acc[it], false, false);
      }
    }
    __syncthreads();
  }

  if (mblk <= 1) {
    // transposed store: yT[b][n][o], 8 contiguous halves per lane -> b128
    _Float16* ysel = (mblk == 0) ? qT : kT;
#pragma unroll
    for (int it = 0; it < 2; ++it) {
      const int t  = wave * 2 + it;
      const int ti = t >> 2, tj = t & 3;
      const int n  = n0 + tj * 16 + ll;
      const int w0 = ti * 16 + 8 * lh;
      v8h pk;
#pragma unroll
      for (int r = 0; r < 8; ++r)
        pk[r] = (_Float16)(acc[it][r] + bsel[w0 + r]);
      *(v8h*)(ysel + ((size_t)b * HW + n) * CQ + w0) = pk;
    }
  } else {
    // V keeps [b][c][pixel] layout (needed as B-matrix for O += P*V^T)
#pragma unroll
    for (int it = 0; it < 2; ++it) {
      const int t  = wave * 2 + it;
      const int ti = t >> 2, tj = t & 3;
      const int n  = n0 + tj * 16 + ll;
#pragma unroll
      for (int r = 0; r < 8; ++r) {
        const int wr = wrow0 + ti * 16 + r + 8 * lh;
        vh[((size_t)b * CIN + wr) * HW + n] = (_Float16)(acc[it][r] + bsel[wr]);
      }
    }
  }
}

// ---------------------------------------------------------------------------
// Kernel 2: flash-style attention. One block per (batch, 64-query tile).
// Q/K tiles arrive via async-to-LDS copies of contiguous 8KB blocks.
// S = QtK via WMMA -> LDS, online softmax (4 lanes/row, shfl quad-reduce),
// O accumulated per wave over a 64-channel slice (4x4 f32 fragments),
// V B-fragments streamed straight from global (L2-resident, contiguous runs).
// ---------------------------------------------------------------------------
__global__ __launch_bounds__(256)
void attention_kernel(const _Float16* __restrict__ qT,
                      const _Float16* __restrict__ kT,
                      const _Float16* __restrict__ vh,
                      const float* __restrict__ alpha,
                      float* __restrict__ out)
{
  __shared__ _Float16 LQ[TILE][72];   // LQ[i][c]
  __shared__ _Float16 LK[TILE][72];   // LK[j][c]
  __shared__ _Float16 LP[TILE][72];   // LP[i][j]  softmax probs (f16)
  __shared__ float    LS[TILE][68];   // LS[i][j]  raw scores
  __shared__ float    rowM[TILE], rowL[TILE], rowA[TILE];

  const int i0   = blockIdx.x * TILE;
  const int b    = blockIdx.y;
  const int tid  = threadIdx.x;
  const int wave = tid >> 5;
  const int lane = tid & 31;
  const int lh   = lane >> 4;
  const int ll   = lane & 15;

  const _Float16* qtile = qT + ((size_t)b * HW + i0) * CQ;   // contiguous 8KB
  const _Float16* kbase = kT + (size_t)b * HW * CQ;
  const _Float16* vbase = vh + (size_t)b * CIN * HW;

  async_copy_tile64(qtile, LQ, tid);               // Q^T tile, staged once
  if (tid < TILE) { rowM[tid] = -1e30f; rowL[tid] = 0.0f; }
  wait_async_all();
  __syncthreads();

  const int c0 = wave * 64;           // this wave's output-channel slice
  v8f acc[4][4];
#pragma unroll
  for (int ti = 0; ti < 4; ++ti)
#pragma unroll
    for (int tc = 0; tc < 4; ++tc)
#pragma unroll
      for (int r = 0; r < 8; ++r) acc[ti][tc][r] = 0.0f;

  for (int j0 = 0; j0 < HW; j0 += TILE) {
    // stage K^T tile asynchronously (contiguous 8KB -> padded LDS)
    async_copy_tile64(kbase + (size_t)j0 * CQ, LK, tid);
    if (j0 + TILE < HW)   // prefetch next K tile into cache hierarchy
      __builtin_prefetch(kbase + (size_t)(j0 + TILE) * CQ + tid * 16, 0, 1);
    wait_async_all();
    __syncthreads();

    // S(64x64) = Q^T K : 16 tiles of 16x16, two per wave, K-dim = Cq = 64
#pragma unroll
    for (int it = 0; it < 2; ++it) {
      const int t  = wave * 2 + it;
      const int ti = t >> 2, tj = t & 3;
      const _Float16* arow = &LQ[ti * 16 + ll][0];
      const _Float16* bcol = &LK[tj * 16 + ll][0];
      v8f s;
#pragma unroll
      for (int r = 0; r < 8; ++r) s[r] = 0.0f;
#pragma unroll
      for (int ks = 0; ks < CQ; ks += 32) {
        v16h af = load_a_frag(arow, ks, lh);
        v16h bf = load_b_frag(bcol, ks, lh);
        s = __builtin_amdgcn_wmma_f32_16x16x32_f16(false, af, false, bf,
                                                   (short)0, s, false, false);
      }
#pragma unroll
      for (int r = 0; r < 8; ++r)
        LS[ti * 16 + r + 8 * lh][tj * 16 + ll] = s[r];
    }
    __syncthreads();

    // online softmax: 4 lanes per row, quad reduction via shfl_xor
    {
      const int i = tid >> 2;
      const int p = tid & 3;
      const float mo = rowM[i];
      const float4* ls4 = (const float4*)&LS[i][p * 16];
      float4 c0v = ls4[0], c1v = ls4[1], c2v = ls4[2], c3v = ls4[3];
      float mx = fmaxf(fmaxf(fmaxf(c0v.x, c0v.y), fmaxf(c0v.z, c0v.w)),
                       fmaxf(fmaxf(c1v.x, c1v.y), fmaxf(c1v.z, c1v.w)));
      mx = fmaxf(mx, fmaxf(fmaxf(fmaxf(c2v.x, c2v.y), fmaxf(c2v.z, c2v.w)),
                           fmaxf(fmaxf(c3v.x, c3v.y), fmaxf(c3v.z, c3v.w))));
      mx = fmaxf(mx, __shfl_xor(mx, 1, 32));
      mx = fmaxf(mx, __shfl_xor(mx, 2, 32));
      const float mn = fmaxf(mo, mx);
      float s = 0.0f;
      const float* lsr = &LS[i][p * 16];
#pragma unroll
      for (int j = 0; j < 16; ++j) {
        const float pe = __expf(lsr[j] - mn);
        LP[i][p * 16 + j] = (_Float16)pe;
        s += pe;
      }
      s += __shfl_xor(s, 1, 32);
      s += __shfl_xor(s, 2, 32);
      if (p == 0) {
        const float sc = __expf(mo - mn);
        rowL[i] = sc * rowL[i] + s;
        rowM[i] = mn;
        rowA[i] = sc;
      }
    }
    __syncthreads();

    // rescale running O accumulators by per-row factor
#pragma unroll
    for (int ti = 0; ti < 4; ++ti)
#pragma unroll
      for (int r = 0; r < 8; ++r) {
        const float sc = rowA[ti * 16 + r + 8 * lh];
#pragma unroll
        for (int tc = 0; tc < 4; ++tc) acc[ti][tc][r] *= sc;
      }

    // O += P @ V^T ; V B-fragments straight from global (j runs contiguous)
#pragma unroll
    for (int ks = 0; ks < TILE; ks += 32) {
      v16h bf[4];
#pragma unroll
      for (int tc = 0; tc < 4; ++tc) {
        const _Float16* vcol = vbase + (size_t)(c0 + tc * 16 + ll) * HW + j0;
        bf[tc] = load_b_frag(vcol, ks, lh);
      }
#pragma unroll
      for (int ti = 0; ti < 4; ++ti) {
        v16h af = load_a_frag(&LP[ti * 16 + ll][0], ks, lh);
#pragma unroll
        for (int tc = 0; tc < 4; ++tc) {
          acc[ti][tc] = __builtin_amdgcn_wmma_f32_16x16x32_f16(false, af, false, bf[tc],
                                                               (short)0, acc[ti][tc],
                                                               false, false);
        }
      }
    }
    __syncthreads();
  }

  // epilogue: out[b][c][i] = alpha * O[i][c] / l[i]
  const float asc = alpha[0];
#pragma unroll
  for (int ti = 0; ti < 4; ++ti) {
    const int il = ti * 16 + 8 * lh;
#pragma unroll
    for (int r = 0; r < 8; ++r) {
      const float inv = asc / rowL[il + r];
      const int i = i0 + il + r;
#pragma unroll
      for (int tc = 0; tc < 4; ++tc) {
        const int c = c0 + tc * 16 + ll;
        out[((size_t)b * CIN + c) * HW + i] = acc[ti][tc][r] * inv;
      }
    }
  }
}

// ---------------------------------------------------------------------------
extern "C" void kernel_launch(void* const* d_in, const int* in_sizes, int n_in,
                              void* d_out, int out_size, void* d_ws, size_t ws_size,
                              hipStream_t stream) {
  const float* x  = (const float*)d_in[0];
  const float* Wq = (const float*)d_in[1];
  const float* bq = (const float*)d_in[2];
  const float* Wk = (const float*)d_in[3];
  const float* bk = (const float*)d_in[4];
  const float* Wv = (const float*)d_in[5];
  const float* bv = (const float*)d_in[6];
  const float* al = (const float*)d_in[7];
  float* out = (float*)d_out;

  _Float16* qT = (_Float16*)d_ws;                      // [4][4096][64]  f16
  _Float16* kT = qT + (size_t)BATCH * HW * CQ;         // [4][4096][64]  f16
  _Float16* vh = kT + (size_t)BATCH * HW * CQ;         // [4][512][4096] f16

  dim3 g1(HW / TILE, (2 * CQ + CIN) / TILE, BATCH);    // (64, 10, 4)
  qkv_proj_kernel<<<g1, 256, 0, stream>>>(x, Wq, bq, Wk, bk, Wv, bv, qT, kT, vh);

  dim3 g2(HW / TILE, BATCH);                           // (64, 4)
  attention_kernel<<<g2, 256, 0, stream>>>(qT, kT, vh, al, out);
}
